// ShiftedWindows_MSA_3D_67499706024010
// MI455X (gfx1250) — compile-verified, matrix operand order checked by reading, not run on hardware
//
#include <hip/hip_runtime.h>
#include <hip/hip_bf16.h>

typedef __attribute__((ext_vector_type(16))) _Float16 v16h;
typedef __attribute__((ext_vector_type(8)))  _Float16 h8;
typedef __attribute__((ext_vector_type(8)))  float    v8f;
typedef __attribute__((ext_vector_type(4)))  float    f4;

#define NTHREADS 256
#define NWAVES   8

// LDS arena layout (bytes). All offsets 16B-aligned.
#define OFF_Q    0          // 112x128 f16 : Q (pre-scaled)
#define OFF_K    28672      // 112x128 f16 : K
#define OFF_VT   57344      // 128x128 f16 : V transposed [ch][pos]
#define OFF_P    90112      // 112x128 f16 : phase A: x window (A matrix); phase B: scores/probs
#define OFF_O    118784     // 112x128 f16 : attention output (all heads)
#define OFF_CNT  147456     // 112 i32    : shift-mask region id per token
#define OFF_ROW  147904     // 112 i32    : global f32 offset of each token's (rolled) position
#define SMEM_BYTES 148352

// ---- fragment loaders -------------------------------------------------------
// 16-bit A/B operand K-layout (ISA 7.12.2): lane holds 16 halves as two
// contiguous 8-element runs: K in [base, base+8) and [base+16, base+24),
// base = (lane<16 ? 0 : 8).  B mirrors A with the tile column in lanes.

__device__ __forceinline__ v16h frag_lds(const _Float16* __restrict__ row,
                                         int cbase, int lane) {
  const int kb = cbase + ((lane & 16) ? 8 : 0);
  h8 a = *(const h8*)(row + kb);
  h8 b = *(const h8*)(row + kb + 16);
  v16h r;
#pragma unroll
  for (int i = 0; i < 8; ++i) { r[i] = a[i]; r[i + 8] = b[i]; }
  return r;
}

// B fragment straight from a row-major fp32 weight matrix W[o][128]:
// B[c][o] = W[o*128 + c].  Two 8-float contiguous runs -> 4x b128 loads.
__device__ __forceinline__ v16h frag_w128(const float* __restrict__ W,
                                          int o, int ks, int lane) {
  const int kb = ks * 32 + ((lane & 16) ? 8 : 0);
  const float* p = W + o * 128 + kb;
  f4 a0 = *(const f4*)(p);
  f4 a1 = *(const f4*)(p + 4);
  f4 b0 = *(const f4*)(p + 16);
  f4 b1 = *(const f4*)(p + 20);
  v16h r;
#pragma unroll
  for (int i = 0; i < 4; ++i) {
    r[i]      = (_Float16)a0[i];
    r[4 + i]  = (_Float16)a1[i];
    r[8 + i]  = (_Float16)b0[i];
    r[12 + i] = (_Float16)b1[i];
  }
  return r;
}

// C/D layout: VGPR r holds (M = m0 + r + (lane<16?0:8), N = n0 + lane%16)
__device__ __forceinline__ void store_tile(_Float16* __restrict__ dst,
                                           int m0, int n0, int lane,
                                           v8f acc, float mul) {
  const int n  = n0 + (lane & 15);
  const int mb = m0 + ((lane & 16) ? 8 : 0);
#pragma unroll
  for (int r = 0; r < 8; ++r)
    dst[(mb + r) * 128 + n] = (_Float16)(acc[r] * mul);
}

__global__ void __launch_bounds__(NTHREADS)
swin3d_msa_kernel(const float* __restrict__ xg,
                  const float* __restrict__ qkvw,   // (384,128)
                  const float* __restrict__ projw,  // (128,128)
                  const float* __restrict__ projb,  // (128)
                  const float* __restrict__ rpb,    // (507,4)
                  float* __restrict__ outg) {
  extern __shared__ char smem[] __attribute__((aligned(16)));
  _Float16* lQ  = (_Float16*)(smem + OFF_Q);
  _Float16* lK  = (_Float16*)(smem + OFF_K);
  _Float16* lVt = (_Float16*)(smem + OFF_VT);
  _Float16* lP  = (_Float16*)(smem + OFF_P);
  _Float16* lX  = lP;  // phase A alias
  _Float16* lO  = (_Float16*)(smem + OFF_O);
  int* lCnt = (int*)(smem + OFF_CNT);
  int* lRow = (int*)(smem + OFF_ROW);

  const int tid  = threadIdx.x;
  const int lane = tid & 31;
  const int wv   = tid >> 5;

  // window decomposition: blocks ordered (b, bt, bh, bw) = (2, 8, 8, 8)
  const int blk = blockIdx.x;
  const int b   = blk >> 9;
  const int wb  = blk & 511;
  const int bt  = wb >> 6;
  const int bh  = (wb >> 3) & 7;
  const int bw  = wb & 7;

  // zero the f16 arena (padding rows/cols must be 0 for the WMMA tiles)
  {
    uint32_t* z = (uint32_t*)smem;
    for (int i = tid; i < (OFF_CNT >> 2); i += NTHREADS) z[i] = 0u;
  }

  // per-token metadata: region id (shift mask) + global row offset.
  // rolled coords g; source/dest position = (g + shift) % dim (input gather
  // and output scatter use the identical mapping).
  if (tid < 98) {
    const int ti = tid / 49;
    const int rm = tid % 49;
    const int hi = rm / 7;
    const int wi = rm % 7;
    const int gT = bt * 2 + ti, gH = bh * 7 + hi, gW = bw * 7 + wi;
    const int rt = (gT < 14) ? 0 : ((gT < 15) ? 1 : 2);
    const int rh = (gH < 49) ? 0 : ((gH < 53) ? 1 : 2);
    const int rw = (gW < 49) ? 0 : ((gW < 53) ? 1 : 2);
    lCnt[tid] = rt * 9 + rh * 3 + rw;
    const int sT = (gT + 1) & 15;
    int sH = gH + 3; if (sH >= 56) sH -= 56;
    int sW = gW + 3; if (sW >= 56) sW -= 56;
    lRow[tid] = (((b * 16 + sT) * 56 + sH) * 56 + sW) * 128;
  }
  __syncthreads();

  // ---- gather window x (98x128 f32 -> f16 in LDS, rows 98..111 stay 0) ----
  for (int idx = tid; idx < 98 * 32; idx += NTHREADS) {
    const int row = idx >> 5;
    const int c4  = (idx & 31) << 2;
    f4 v = *(const f4*)(xg + lRow[row] + c4);
    _Float16* d = lX + row * 128 + c4;
    d[0] = (_Float16)v[0]; d[1] = (_Float16)v[1];
    d[2] = (_Float16)v[2]; d[3] = (_Float16)v[3];
  }
  __syncthreads();

  // ---- QKV GEMM: (112x384) = X(112x128) @ qkv_w^T(128x384) ----------------
  const float SCALE = 0.17677669529663687f;  // 32^-0.5
  for (int tile = wv; tile < 7 * 24; tile += NWAVES) {
    const int mt = tile / 24;
    const int nt = tile % 24;
    const int o0 = nt * 16;
    const _Float16* arow = lX + (mt * 16 + (lane & 15)) * 128;
    const int o = o0 + (lane & 15);
    v8f acc = {};
#pragma unroll
    for (int ks = 0; ks < 4; ++ks) {
      v16h a  = frag_lds(arow, ks * 32, lane);
      v16h bb = frag_w128(qkvw, o, ks, lane);
      acc = __builtin_amdgcn_wmma_f32_16x16x32_f16(false, a, false, bb,
                                                   (short)0, acc, false, false);
    }
    const int m0 = mt * 16;
    if (o0 < 128) {
      store_tile(lQ, m0, o0, lane, acc, SCALE);          // q * scale
    } else if (o0 < 256) {
      store_tile(lK, m0, o0 - 128, lane, acc, 1.0f);     // k
    } else {                                             // v, transposed
      const int ch = (o0 - 256) + (lane & 15);
      const int mb = m0 + ((lane & 16) ? 8 : 0);
#pragma unroll
      for (int r = 0; r < 8; ++r)
        lVt[ch * 128 + mb + r] = (_Float16)acc[r];
    }
  }
  __syncthreads();

  // ---- attention, one head at a time --------------------------------------
  for (int h = 0; h < 4; ++h) {
    // S = Q_h @ K_h^T : K dim = hd = 32 -> exactly one WMMA per 16x16 tile
    for (int tile = wv; tile < 49; tile += NWAVES) {
      const int mt = tile / 7, nt = tile % 7;
      const _Float16* arow = lQ + (mt * 16 + (lane & 15)) * 128;
      const _Float16* brow = lK + (nt * 16 + (lane & 15)) * 128;
      v16h a  = frag_lds(arow, h * 32, lane);
      v16h bb = frag_lds(brow, h * 32, lane);
      v8f acc = {};
      acc = __builtin_amdgcn_wmma_f32_16x16x32_f16(false, a, false, bb,
                                                   (short)0, acc, false, false);
      store_tile(lP, mt * 16, nt * 16, lane, acc, 1.0f);
    }
    __syncthreads();

    // row softmax with on-the-fly rel-pos bias + shift mask
    if (tid < 98) {
      const int tr = tid / 49;
      const int rr = tid % 49;
      const int hr = rr / 7;
      const int wr = rr % 7;
      const int cr = lCnt[tid];
      _Float16* row = lP + tid * 128;
      float vmax = -3.0e38f;
      int tm = 0, hm = 0, wm = 0;
      for (int m = 0; m < 98; ++m) {
        const int rpi = (tr - tm + 1) * 169 + (hr - hm + 6) * 13 + (wr - wm + 6);
        float s = (float)row[m] + rpb[rpi * 4 + h];
        if (lCnt[m] != cr) s -= 100.0f;
        vmax = fmaxf(vmax, s);
        if (++wm == 7) { wm = 0; if (++hm == 7) { hm = 0; ++tm; } }
      }
      float sum = 0.0f; tm = hm = wm = 0;
      for (int m = 0; m < 98; ++m) {
        const int rpi = (tr - tm + 1) * 169 + (hr - hm + 6) * 13 + (wr - wm + 6);
        float s = (float)row[m] + rpb[rpi * 4 + h];
        if (lCnt[m] != cr) s -= 100.0f;
        const float e = __expf(s - vmax);
        sum += e;
        row[m] = (_Float16)e;
        if (++wm == 7) { wm = 0; if (++hm == 7) { hm = 0; ++tm; } }
      }
      const float inv = 1.0f / sum;
      for (int m = 0; m < 98; ++m) row[m] = (_Float16)((float)row[m] * inv);
      for (int m = 98; m < 112; ++m) row[m] = (_Float16)0.0f;  // K-padding
    } else if (tid < 112) {
      _Float16* row = lP + tid * 128;
      for (int m = 0; m < 112; ++m) row[m] = (_Float16)0.0f;   // junk rows
    }
    __syncthreads();

    // O_h(112x32) = P(112x128) @ V_h(128x32); B fragments from V^T rows
    for (int tile = wv; tile < 14; tile += NWAVES) {
      const int mt  = tile % 7;
      const int ntc = tile / 7;
      const _Float16* arow = lP + (mt * 16 + (lane & 15)) * 128;
      const _Float16* brow = lVt + (h * 32 + ntc * 16 + (lane & 15)) * 128;
      v8f acc = {};
#pragma unroll
      for (int ks = 0; ks < 4; ++ks) {
        v16h a  = frag_lds(arow, ks * 32, lane);
        v16h bb = frag_lds(brow, ks * 32, lane);
        acc = __builtin_amdgcn_wmma_f32_16x16x32_f16(false, a, false, bb,
                                                     (short)0, acc, false, false);
      }
      store_tile(lO, mt * 16, h * 32 + ntc * 16, lane, acc, 1.0f);
    }
    __syncthreads();
  }

  // ---- proj GEMM + bias, scatter to rolled-back positions -----------------
  for (int tile = wv; tile < 56; tile += NWAVES) {
    const int mt = tile >> 3;
    const int nt = tile & 7;
    const _Float16* arow = lO + (mt * 16 + (lane & 15)) * 128;
    const int o = nt * 16 + (lane & 15);
    v8f acc = {};
#pragma unroll
    for (int ks = 0; ks < 4; ++ks) {
      v16h a  = frag_lds(arow, ks * 32, lane);
      v16h bb = frag_w128(projw, o, ks, lane);
      acc = __builtin_amdgcn_wmma_f32_16x16x32_f16(false, a, false, bb,
                                                   (short)0, acc, false, false);
    }
    const float pb = projb[o];
    const int mb = mt * 16 + ((lane & 16) ? 8 : 0);
#pragma unroll
    for (int r = 0; r < 8; ++r) {
      const int m = mb + r;
      if (m < 98) outg[lRow[m] + o] = acc[r] + pb;
    }
  }
}

extern "C" void kernel_launch(void* const* d_in, const int* in_sizes, int n_in,
                              void* d_out, int out_size, void* d_ws, size_t ws_size,
                              hipStream_t stream) {
  (void)in_sizes; (void)n_in; (void)d_ws; (void)ws_size; (void)out_size;
  const float* x     = (const float*)d_in[0];
  const float* qkvw  = (const float*)d_in[1];
  const float* projw = (const float*)d_in[2];
  const float* projb = (const float*)d_in[3];
  const float* rpb   = (const float*)d_in[4];
  float* out = (float*)d_out;

  // allow >64KB dynamic LDS (WGP has 320KB); harmless if already set
  (void)hipFuncSetAttribute(reinterpret_cast<const void*>(swin3d_msa_kernel),
                            hipFuncAttributeMaxDynamicSharedMemorySize,
                            SMEM_BYTES);

  swin3d_msa_kernel<<<1024, NTHREADS, SMEM_BYTES, stream>>>(
      x, qkvw, projw, projb, rpb, out);
}
